// Head_85770496901233
// MI455X (gfx1250) — matrix operand order repaired and checked by reading, not compiled, not run on hardware
//
#include <hip/hip_runtime.h>

typedef _Float16 v16h __attribute__((ext_vector_type(16)));
typedef _Float16 v8h  __attribute__((ext_vector_type(8)));
typedef _Float16 v4h  __attribute__((ext_vector_type(4)));
typedef float    v8f  __attribute__((ext_vector_type(8)));

#define HS   64
#define CDIM 1024
#define TSEQ 2048
#define NB   4

// ---- fragment builders -----------------------------------------------------
// A-frag (16x32 f16, MxK): lane holds row (lane&15); elements [0..7] = K kb..kb+7,
// elements [8..15] = K 16+kb..16+kb+7, kb = (lane<16)?0:8.  base = rowptr + c0 + kb.
__device__ __forceinline__ v16h a_frag_from_f32(const float* base) {
    v8f lo = *(const v8f*)(base);
    v8f hi = *(const v8f*)(base + 16);
    v16h a;
#pragma unroll
    for (int i = 0; i < 8; ++i) { a[i] = (_Float16)lo[i]; a[8 + i] = (_Float16)hi[i]; }
    return a;
}
__device__ __forceinline__ v16h a_frag_from_f16(const _Float16* base) {
    v8h lo = *(const v8h*)(base);
    v8h hi = *(const v8h*)(base + 16);
    v16h a;
#pragma unroll
    for (int i = 0; i < 8; ++i) { a[i] = lo[i]; a[8 + i] = hi[i]; }
    return a;
}
// B-frag (32x16 f16, KxN): lane holds column (lane&15); elements [0..7] = K kb..kb+7,
// elements [8..15] = K 16+kb..16+kb+7, kb = (lane<16)?0:8.  base = colptr + c0 + kbB.
__device__ __forceinline__ v16h b_frag_from_f16(const _Float16* base) {
    v8h lo = *(const v8h*)(base);
    v8h hi = *(const v8h*)(base + 16);
    v16h b;
#pragma unroll
    for (int i = 0; i < 8; ++i) { b[i] = lo[i]; b[8 + i] = hi[i]; }
    return b;
}

// ---- Phase 0: one-shot fp32 -> f16 conversion of Wq|Wk|Wv ------------------
__global__ __launch_bounds__(256) void cvt_w(
    const float* __restrict__ Wq, const float* __restrict__ Wk,
    const float* __restrict__ Wv, _Float16* __restrict__ Wh)
{
    const int n = HS * CDIM;                       // 65536 per matrix
    const int i = (blockIdx.x * 256 + threadIdx.x) * 4;   // [0, 3n)
    const float* src = (i < n) ? (Wq + i) : (i < 2 * n ? (Wk + (i - n))
                                                       : (Wv + (i - 2 * n)));
    const float4 v = *(const float4*)src;
    v4h h;
    h[0] = (_Float16)v.x; h[1] = (_Float16)v.y;
    h[2] = (_Float16)v.z; h[3] = (_Float16)v.w;
    *(v4h*)(Wh + i) = h;
}

// ---- Phase 1: q/k/v projection, x fp32 -> f16 frags, W read as f16 ---------
__global__ __launch_bounds__(256) void proj_qkv(
    const float* __restrict__ x, const _Float16* __restrict__ Wh,
    _Float16* __restrict__ qh, _Float16* __restrict__ kh,
    _Float16* __restrict__ vth)
{
    const int lane = threadIdx.x & 31;
    const int wave = threadIdx.x >> 5;
    const int row0 = blockIdx.x * 128 + wave * 16;      // row in [0, B*T)
    const int nn   = lane & 15;
    const int kbA  = (lane < 16) ? 0 : 8;
    const int kbB  = (lane < 16) ? 0 : 8;
    const int lrow = row0 + nn;

    v8f zero = {};
    v8f acc[3][4];
#pragma unroll
    for (int w = 0; w < 3; ++w)
#pragma unroll
        for (int ht = 0; ht < 4; ++ht) acc[w][ht] = zero;

    for (int c0 = 0; c0 < CDIM; c0 += 32) {
        v16h a = a_frag_from_f32(x + (size_t)lrow * CDIM + c0 + kbA);
#pragma unroll
        for (int w = 0; w < 3; ++w) {
#pragma unroll
            for (int ht = 0; ht < 4; ++ht) {
                const _Float16* wr =
                    Wh + (size_t)w * HS * CDIM + (size_t)(ht * 16 + nn) * CDIM + c0 + kbB;
                acc[w][ht] = __builtin_amdgcn_wmma_f32_16x16x32_f16(
                    false, a, false, b_frag_from_f16(wr), (short)0, acc[w][ht],
                    false, false);
            }
        }
    }

    const int bb = row0 / TSEQ;
    const int t0 = row0 % TSEQ;
    const int mh = 8 * (lane >> 4);
#pragma unroll
    for (int ht = 0; ht < 4; ++ht) {
        const int h = ht * 16 + nn;
#pragma unroll
        for (int r = 0; r < 8; ++r) {
            qh[(size_t)(row0 + r + mh) * HS + h] = (_Float16)acc[0][ht][r];
            kh[(size_t)(row0 + r + mh) * HS + h] = (_Float16)acc[1][ht][r];
        }
        // transposed V: lane's 8 values are contiguous in t -> one b128 store
        v8h vv;
#pragma unroll
        for (int r = 0; r < 8; ++r) vv[r] = (_Float16)acc[2][ht][r];
        *(v8h*)(vth + ((size_t)bb * HS + h) * TSEQ + t0 + mh) = vv;
    }
}

// ---- Phase 2: flash attention, one 16-query tile per wave ------------------
__global__ __launch_bounds__(128) void attn_head(
    const _Float16* __restrict__ qh, const _Float16* __restrict__ kh,
    const _Float16* __restrict__ vth, float* __restrict__ out)
{
    __shared__ __align__(16) _Float16 Plds[4][16 * 32];   // wave-private P tiles

    const int lane  = threadIdx.x & 31;
    const int wave  = threadIdx.x >> 5;
    const int grow0 = blockIdx.x * 64 + wave * 16;        // global query row base
    const int bb    = grow0 / TSEQ;
    const int t0    = grow0 % TSEQ;
    const int nn    = lane & 15;
    const int mh    = 8 * (lane >> 4);
    const int kbA   = (lane < 16) ? 0 : 8;
    const int kbB   = (lane < 16) ? 0 : 8;

    // Q A-fragments for the whole tile (K = head dim 64 -> 2 frags)
    v16h aq[2];
    {
        const _Float16* qr = qh + (size_t)(grow0 + nn) * HS;
        aq[0] = a_frag_from_f16(qr + 0 + kbA);
        aq[1] = a_frag_from_f16(qr + 32 + kbA);
    }

    v8f zero = {};
    v8f O[4];
    float M[8], L[8];
#pragma unroll
    for (int ht = 0; ht < 4; ++ht) O[ht] = zero;
#pragma unroll
    for (int r = 0; r < 8; ++r) { M[r] = -3.0e38f; L[r] = 0.0f; }

    const int jmax = t0 + 15;                    // last needed key index
    _Float16* pl = &Plds[wave][0];

    for (int j0 = 0; j0 <= jmax; j0 += 32) {
        // prefetch next K tile (speculative; OOB silently dropped)
        __builtin_prefetch(kh + ((size_t)bb * TSEQ + j0 + 32 + nn) * HS, 0, 1);

        float Sv[2][8];
#pragma unroll
        for (int jt = 0; jt < 2; ++jt) {
            const int jb = j0 + 16 * jt;
            if (jb <= jmax) {                    // wave-uniform branch: EXEC all-1s
                const _Float16* kr = kh + ((size_t)bb * TSEQ + jb + nn) * HS;
                v8f s = zero;
                s = __builtin_amdgcn_wmma_f32_16x16x32_f16(
                        false, aq[0], false, b_frag_from_f16(kr + 0 + kbB),
                        (short)0, s, false, false);
                s = __builtin_amdgcn_wmma_f32_16x16x32_f16(
                        false, aq[1], false, b_frag_from_f16(kr + 32 + kbB),
                        (short)0, s, false, false);
#pragma unroll
                for (int r = 0; r < 8; ++r) {
                    const int qi = t0 + r + mh;
                    const int ki = jb + nn;
                    Sv[jt][r] = (ki <= qi) ? s[r] * 0.125f : -3.0e38f;
                }
            } else {
#pragma unroll
                for (int r = 0; r < 8; ++r) Sv[jt][r] = -3.0e38f;
            }
        }

        // online softmax: per-row (16-lane half-wave) max / sum
        float alpha[8];
#pragma unroll
        for (int r = 0; r < 8; ++r) {
            float mx = fmaxf(Sv[0][r], Sv[1][r]);
#pragma unroll
            for (int off = 1; off < 16; off <<= 1)
                mx = fmaxf(mx, __shfl_xor(mx, off, 32));
            const float Mn = fmaxf(M[r], mx);
            alpha[r] = __expf(M[r] - Mn);
            M[r] = Mn;
            float rs = 0.0f;
#pragma unroll
            for (int jt = 0; jt < 2; ++jt) {
                const float p = __expf(Sv[jt][r] - Mn);
                Sv[jt][r] = p;
                rs += p;
            }
#pragma unroll
            for (int off = 1; off < 16; off <<= 1)
                rs += __shfl_xor(rs, off, 32);
            L[r] = L[r] * alpha[r] + rs;
        }
#pragma unroll
        for (int ht = 0; ht < 4; ++ht)
#pragma unroll
            for (int r = 0; r < 8; ++r) O[ht][r] *= alpha[r];

        // C/D-layout P -> A-layout via wave-private LDS tile
#pragma unroll
        for (int jt = 0; jt < 2; ++jt)
#pragma unroll
            for (int r = 0; r < 8; ++r)
                pl[(r + mh) * 32 + jt * 16 + nn] = (_Float16)Sv[jt][r];

        asm volatile("s_wait_dscnt 0x0" ::: "memory");
        __builtin_amdgcn_wave_barrier();

        v16h aP = a_frag_from_f16(pl + nn * 32 + kbA);

        // O += P (16x32) x V (32x64), V read as rows of vT (contiguous)
#pragma unroll
        for (int ht = 0; ht < 4; ++ht) {
            const _Float16* vr =
                vth + ((size_t)bb * HS + ht * 16 + nn) * TSEQ + j0 + kbB;
            O[ht] = __builtin_amdgcn_wmma_f32_16x16x32_f16(
                false, aP, false, b_frag_from_f16(vr), (short)0, O[ht], false, false);
        }

        // ensure P loads retired before next iteration's stores (WAR via LDS)
        asm volatile("s_wait_dscnt 0x0" ::: "memory");
        __builtin_amdgcn_wave_barrier();
    }

    // normalize and store fp32 output [B,T,H]
#pragma unroll
    for (int ht = 0; ht < 4; ++ht) {
#pragma unroll
        for (int r = 0; r < 8; ++r) {
            out[(size_t)(grow0 + r + mh) * HS + ht * 16 + nn] = O[ht][r] / L[r];
        }
    }
}

// ---- launch ----------------------------------------------------------------
extern "C" void kernel_launch(void* const* d_in, const int* in_sizes, int n_in,
                              void* d_out, int out_size, void* d_ws, size_t ws_size,
                              hipStream_t stream) {
    const float* x  = (const float*)d_in[0];
    const float* Wq = (const float*)d_in[1];
    const float* Wk = (const float*)d_in[2];
    const float* Wv = (const float*)d_in[3];
    float* out = (float*)d_out;

    const size_t n = (size_t)NB * TSEQ * HS;        // 512K elements each
    _Float16* vth = (_Float16*)d_ws;                // [B][H][T]
    _Float16* kh  = vth + n;                        // [B*T][H]
    _Float16* qh  = kh + n;                         // [B*T][H]
    _Float16* Wh  = qh + n;                         // [3][H][C] f16

    cvt_w<<<(3 * HS * CDIM) / (256 * 4), 256, 0, stream>>>(Wq, Wk, Wv, Wh);
    proj_qkv<<<(NB * TSEQ) / 128, 256, 0, stream>>>(x, Wh, qh, kh, vth);
    attn_head<<<(NB * TSEQ) / 64, 128, 0, stream>>>(qh, kh, vth, out);
}
